// LcaModule_84095459656336
// MI455X (gfx1250) — matrix operand order repaired and checked by reading, not compiled, not run on hardware
//
#include <hip/hip_runtime.h>

// ---------------------------------------------------------------------------
// LCA sparse coding on MI455X (gfx1250), wave32 + v_wmma_f32_16x16x32_bf16.
// B=16384 batch, D=784 input dim (padded to KP=800), L=1024 latents.
// ---------------------------------------------------------------------------

#define BATCH   16384
#define DIM     784
#define KP      800          // 784 padded up to a multiple of 32 (25 chunks)
#define LAT     1024
#define NSTEPS  49           // NUM_STEPS - 1 scan steps
#define LAMBDA  0.3f
#define ETA     0.1f
#define MROWS   32           // batch rows per LCA block (2 M-tiles)

typedef __attribute__((ext_vector_type(16))) __bf16       bf16x16;
typedef __attribute__((ext_vector_type(8)))  float        floatx8;
typedef __attribute__((ext_vector_type(4)))  unsigned int uintx4;

union Frag { uintx4 u[2]; bf16x16 v; };

__device__ __forceinline__ float thr(float u) {
    return u > LAMBDA ? u - LAMBDA : 0.0f;
}

// A-fragment (16x32 bf16) from a row-major global matrix, leading dim ld.
// ISA layout: lanes 0-15 hold M=lane, K 0-7 & 16-23; lanes 16-31 hold K 8-15 & 24-31.
__device__ __forceinline__ bf16x16
load_a_frag_global(const __bf16* base, int ld, int m0, int k0, int lane) {
    const int row = m0 + (lane & 15);
    const int sel = (lane >> 4) ? 8 : 0;
    const __bf16* p = base + (size_t)row * ld + k0 + sel;
    Frag f;
    f.u[0] = *(const uintx4*)(p);        // 8 bf16 = 16B
    f.u[1] = *(const uintx4*)(p + 16);   // K+16 block
    return f.v;
}

// Same A-fragment but sourced from LDS (ds_read_b128 x2).
__device__ __forceinline__ bf16x16
load_a_frag_lds(const __bf16* sh, int ld, int k0, int lane) {
    const int row = lane & 15;
    const int sel = (lane >> 4) ? 8 : 0;
    const __bf16* p = sh + row * ld + k0 + sel;
    Frag f;
    f.u[0] = *(const uintx4*)(p);
    f.u[1] = *(const uintx4*)(p + 16);
    return f.v;
}

// B-fragment (32x16 bf16) read from B-TRANSPOSED row-major storage.
// lane = column n0+(lane&15); lanes 0-15 hold K 0-7 & 16-23, lanes >=16
// hold K 8-15 & 24-31.  Two 16B loads.
__device__ __forceinline__ bf16x16
load_b_frag_bt(const __bf16* bt, int ld, int n0, int k0, int lane) {
    const int row  = n0 + (lane & 15);
    const int koff = k0 + ((lane >> 4) << 3);
    const __bf16* p = bt + (size_t)row * ld + koff;
    Frag f;
    f.u[0] = *(const uintx4*)(p);
    f.u[1] = *(const uintx4*)(p + 16);
    return f.v;
}

// ---------------------------------------------------------------------------
// Conversion kernels: f32 -> bf16 with zero K-padding; wT stored as well.
// ---------------------------------------------------------------------------
__global__ void cvt_x_kernel(const float* __restrict__ x, __bf16* __restrict__ xb) {
    const int row = blockIdx.x;
    for (int k = threadIdx.x; k < KP; k += 256) {
        float v = (k < DIM) ? x[(size_t)row * DIM + k] : 0.0f;
        xb[(size_t)row * KP + k] = (__bf16)v;
    }
}

__global__ void cvt_w_kernel(const float* __restrict__ w,
                             __bf16* __restrict__ wb,   // [KP][LAT]
                             __bf16* __restrict__ wtb)  // [LAT][KP]
{
    const int d = blockIdx.x;              // 0..KP-1
    for (int l = threadIdx.x; l < LAT; l += 256) {
        float v = (d < DIM) ? w[(size_t)d * LAT + l] : 0.0f;
        wb[(size_t)d * LAT + l]  = (__bf16)v;
        wtb[(size_t)l * KP + d]  = (__bf16)v;
    }
}

// ---------------------------------------------------------------------------
// g = wT @ w - I  -> bf16 [LAT][LAT]  (symmetric, so row-major == gT)
// One 16x16 tile per wave; 8 waves / block; 64x64 tiles.
// ---------------------------------------------------------------------------
__global__ __launch_bounds__(256) void gemm_g_kernel(const __bf16* __restrict__ wtb,
                                                     __bf16* __restrict__ g) {
    const int lane = threadIdx.x & 31;
    const int tile = blockIdx.x * 8 + (threadIdx.x >> 5);
    const int m0 = (tile >> 6) * 16;
    const int n0 = (tile & 63) * 16;

    floatx8 acc = {0.f,0.f,0.f,0.f,0.f,0.f,0.f,0.f};
    #pragma unroll 5
    for (int kc = 0; kc < KP / 32; ++kc) {
        bf16x16 af = load_a_frag_global(wtb, KP, m0, kc * 32, lane);
        bf16x16 bf = load_b_frag_bt   (wtb, KP, n0, kc * 32, lane);
        acc = __builtin_amdgcn_wmma_f32_16x16x32_bf16(false, af, false, bf,
                                                      (short)0, acc, false, false);
    }
    const int mq = (lane >> 4) << 3;
    const int nl = lane & 15;
    #pragma unroll
    for (int v = 0; v < 8; ++v) {
        const int m = m0 + v + mq, n = n0 + nl;
        float d = acc[v] - ((m == n) ? 1.0f : 0.0f);
        g[(size_t)m * LAT + n] = (__bf16)d;
    }
}

// ---------------------------------------------------------------------------
// b = x @ w  -> f32 [BATCH][LAT]   (A = x_bf16, Bt = wT_bf16)
// ---------------------------------------------------------------------------
__global__ __launch_bounds__(256) void gemm_b_kernel(const __bf16* __restrict__ xb,
                                                     const __bf16* __restrict__ wtb,
                                                     float* __restrict__ b) {
    const int lane = threadIdx.x & 31;
    const int tile = blockIdx.x * 8 + (threadIdx.x >> 5);
    const int m0 = (tile >> 6) * 16;     // 1024 M-tiles
    const int n0 = (tile & 63) * 16;     // 64 N-tiles

    floatx8 acc = {0.f,0.f,0.f,0.f,0.f,0.f,0.f,0.f};
    #pragma unroll 5
    for (int kc = 0; kc < KP / 32; ++kc) {
        bf16x16 af = load_a_frag_global(xb,  KP, m0, kc * 32, lane);
        bf16x16 bf = load_b_frag_bt   (wtb, KP, n0, kc * 32, lane);
        acc = __builtin_amdgcn_wmma_f32_16x16x32_bf16(false, af, false, bf,
                                                      (short)0, acc, false, false);
    }
    const int mq = (lane >> 4) << 3;
    const int nl = lane & 15;
    #pragma unroll
    for (int v = 0; v < 8; ++v)
        b[(size_t)(m0 + v + mq) * LAT + n0 + nl] = acc[v];
}

// ---------------------------------------------------------------------------
// LCA fixed-point iteration.  One block = 512 threads (16 waves) owning 32
// batch rows (2 M-tiles); each wave owns a 64-column slab (2 Mt x 4 Nt) of u
// in registers (WMMA C-layout).  Per step:
//   phase 1: a = relu(u - lambda) -> bf16 LDS staging tile (32 x 1024)
//   phase 2: ag = a@g via WMMA, as a REAL loop over 16 K-pairs (bounded
//            scheduler window => bounded register pressure, no spills).
//            A-fragments cached in VGPRs per K-pair and reused across the
//            4 N-tiles (256B LDS per WMMA); each B-fragment of symmetric g
//            feeds BOTH M-tiles (2x less L2 traffic).
//   u += eta*(b - ag - u) in f32 registers; b re-read from L2 each step.
// Loop-carried registers: u 64 + acc 64 + af 32 + addressing — well under
// the 256-VGPR cap required for 16 waves/WGP (4/SIMD) residency.
// ---------------------------------------------------------------------------
#define A_LD (LAT + 8)   // LDS row pitch in bf16 elems (16B pad, keeps 16B align)

__global__ __launch_bounds__(512) void lca_kernel(const float*  __restrict__ b,
                                                  const __bf16* __restrict__ g,
                                                  __bf16*       __restrict__ a_out) {
    __shared__ __bf16 a_sh[MROWS][A_LD];

    const int lane = threadIdx.x & 31;
    const int wave = threadIdx.x >> 5;    // 0..15
    const int m0   = blockIdx.x * MROWS;
    const int nc   = wave * 64;           // this wave's 64-column slab
    const int mq   = (lane >> 4) << 3;
    const int nl   = lane & 15;

    floatx8 u[2][4];
    #pragma unroll
    for (int mt = 0; mt < 2; ++mt)
        #pragma unroll
        for (int t = 0; t < 4; ++t)
            #pragma unroll
            for (int v = 0; v < 8; ++v)
                u[mt][t][v] = 0.0f;

    #pragma unroll 1
    for (int step = 0; step < NSTEPS; ++step) {
        // phase 1: stage a = thr(u) into LDS (each element owned by one wave)
        #pragma unroll
        for (int mt = 0; mt < 2; ++mt)
            #pragma unroll
            for (int t = 0; t < 4; ++t)
                #pragma unroll
                for (int v = 0; v < 8; ++v)
                    a_sh[mt*16 + v + mq][nc + t*16 + nl] = (__bf16)thr(u[mt][t][v]);
        __syncthreads();

        // phase 2: ag = a @ g with K-pair register-cached A-fragments
        floatx8 acc[2][4];
        #pragma unroll
        for (int mt = 0; mt < 2; ++mt)
            #pragma unroll
            for (int t = 0; t < 4; ++t)
                #pragma unroll
                for (int v = 0; v < 8; ++v)
                    acc[mt][t][v] = 0.0f;

        #pragma unroll 1
        for (int ko = 0; ko < 16; ++ko) {         // 16 K-pairs of 2 chunks
            bf16x16 af[2][2];
            #pragma unroll
            for (int mt = 0; mt < 2; ++mt)
                #pragma unroll
                for (int j = 0; j < 2; ++j)
                    af[mt][j] = load_a_frag_lds(&a_sh[mt*16][0], A_LD,
                                                (ko*2 + j) * 32, lane);
            #pragma unroll
            for (int t = 0; t < 4; ++t) {
                const int n0 = nc + t * 16;
                #pragma unroll
                for (int j = 0; j < 2; ++j) {
                    bf16x16 bf = load_b_frag_bt(g, LAT, n0, (ko*2 + j) * 32, lane);
                    acc[0][t] = __builtin_amdgcn_wmma_f32_16x16x32_bf16(
                        false, af[0][j], false, bf, (short)0, acc[0][t], false, false);
                    acc[1][t] = __builtin_amdgcn_wmma_f32_16x16x32_bf16(
                        false, af[1][j], false, bf, (short)0, acc[1][t], false, false);
                }
            }
        }

        // u update; the excitatory drive b is L2-resident, re-read each step
        #pragma unroll
        for (int mt = 0; mt < 2; ++mt)
            #pragma unroll
            for (int t = 0; t < 4; ++t)
                #pragma unroll
                for (int v = 0; v < 8; ++v) {
                    const float bv =
                        b[(size_t)(m0 + mt*16 + v + mq) * LAT + nc + t*16 + nl];
                    u[mt][t][v] += ETA * (bv - acc[mt][t][v] - u[mt][t][v]);
                }
        __syncthreads();
    }

    // final activation -> bf16 for the reconstruction GEMM
    #pragma unroll
    for (int mt = 0; mt < 2; ++mt)
        #pragma unroll
        for (int t = 0; t < 4; ++t)
            #pragma unroll
            for (int v = 0; v < 8; ++v)
                a_out[(size_t)(m0 + mt*16 + v + mq) * LAT + nc + t*16 + nl] =
                    (__bf16)thr(u[mt][t][v]);
}

// ---------------------------------------------------------------------------
// recon = a @ wT  -> f32 [BATCH][DIM]   (A = a_bf16, Bt = w_bf16 rows 0..783)
// ---------------------------------------------------------------------------
__global__ __launch_bounds__(256) void gemm_recon_kernel(const __bf16* __restrict__ ab,
                                                         const __bf16* __restrict__ wb,
                                                         float* __restrict__ out) {
    const int lane = threadIdx.x & 31;
    const int tile = blockIdx.x * 8 + (threadIdx.x >> 5);
    const int mt = tile / 49, nt = tile % 49;         // 1024 x 49 tiles
    const int m0 = mt * 16,   n0 = nt * 16;

    floatx8 acc = {0.f,0.f,0.f,0.f,0.f,0.f,0.f,0.f};
    #pragma unroll 8
    for (int kc = 0; kc < LAT / 32; ++kc) {
        bf16x16 af = load_a_frag_global(ab, LAT, m0, kc * 32, lane);
        bf16x16 bf = load_b_frag_bt   (wb, LAT, n0, kc * 32, lane);
        acc = __builtin_amdgcn_wmma_f32_16x16x32_bf16(false, af, false, bf,
                                                      (short)0, acc, false, false);
    }
    const int mq = (lane >> 4) << 3;
    const int nl = lane & 15;
    #pragma unroll
    for (int v = 0; v < 8; ++v)
        out[(size_t)(m0 + v + mq) * DIM + n0 + nl] = acc[v];
}

// ---------------------------------------------------------------------------
// Host launcher.  Workspace layout (all 16B aligned):
//   x_bf16 [16384][800] | w_bf16 [800][1024] | wT_bf16 [1024][800]
//   g_bf16 [1024][1024] | a_bf16 [16384][1024] | b_f32 [16384][1024]
// total ~132 MB.
// ---------------------------------------------------------------------------
extern "C" void kernel_launch(void* const* d_in, const int* in_sizes, int n_in,
                              void* d_out, int out_size, void* d_ws, size_t ws_size,
                              hipStream_t stream) {
    const float* x = (const float*)d_in[0];   // [16384, 784]
    const float* w = (const float*)d_in[1];   // [784, 1024]
    float* out = (float*)d_out;               // [16384, 784]

    __bf16* x_bf  = (__bf16*)d_ws;
    __bf16* w_bf  = x_bf  + (size_t)BATCH * KP;
    __bf16* wT_bf = w_bf  + (size_t)KP * LAT;
    __bf16* g_bf  = wT_bf + (size_t)LAT * KP;
    __bf16* a_bf  = g_bf  + (size_t)LAT * LAT;
    float*  b_f   = (float*)(a_bf + (size_t)BATCH * LAT);

    cvt_x_kernel      <<<BATCH, 256, 0, stream>>>(x, x_bf);
    cvt_w_kernel      <<<KP,    256, 0, stream>>>(w, w_bf, wT_bf);
    gemm_g_kernel     <<<(64 * 64) / 8,      256, 0, stream>>>(wT_bf, g_bf);
    gemm_b_kernel     <<<(1024 * 64) / 8,    256, 0, stream>>>(x_bf, wT_bf, b_f);
    lca_kernel        <<<BATCH / MROWS,      512, 0, stream>>>(b_f, g_bf, a_bf);
    gemm_recon_kernel <<<(1024 * 49) / 8,    256, 0, stream>>>(a_bf, w_bf, out);
}
